// DGCNN_20624432956311
// MI455X (gfx1250) — matrix operand order repaired and checked by reading, not compiled, not run on hardware
//
#include <hip/hip_runtime.h>
#include <hip/hip_bf16.h>

#define BATCH 16
#define NPTS  2048
#define KNN   20
#define EPSV  1e-5f

typedef unsigned short u16;
typedef __attribute__((ext_vector_type(2)))  float          v2f;
typedef __attribute__((ext_vector_type(8)))  float          v8f;
typedef __attribute__((ext_vector_type(8)))  unsigned       v8u;
typedef __attribute__((ext_vector_type(16))) __bf16         v16bf;
typedef __attribute__((ext_vector_type(16))) unsigned short v16u;

__device__ __forceinline__ u16 f2bf(float f) {
    union { float f; unsigned u; } v; v.f = f;
    unsigned r = v.u + 0x7FFFu + ((v.u >> 16) & 1u);   // RNE fp32 -> bf16
    return (u16)(r >> 16);
}
// order-preserving float <-> uint map (for LDS atomicMax on floats)
__device__ __forceinline__ unsigned fmap(float f) {
    union { float f; unsigned u; } v; v.f = f;
    return (v.u & 0x80000000u) ? ~v.u : (v.u | 0x80000000u);
}
__device__ __forceinline__ float funmap(unsigned u) {
    union { float f; unsigned u; } v;
    v.u = (u & 0x80000000u) ? (u & 0x7FFFFFFFu) : ~u;
    return v.f;
}

// ------- layer-1 zero-pad: x(B,3,N) -> xpad(B,16,N) fp32 + bf16 (B,N,16) ---
__global__ void k_padx(const float* __restrict__ x, float* __restrict__ xpad,
                       u16* __restrict__ xpadhT) {
    long t = (long)blockIdx.x * blockDim.x + threadIdx.x;
    if (t >= (long)BATCH * 16 * NPTS) return;
    int n = (int)(t % NPTS); long r = t / NPTS;
    int c = (int)(r % 16);   int b = (int)(r / 16);
    float v = (c < 3) ? x[((long)b * 3 + c) * NPTS + n] : 0.f;
    xpad[t] = v;
    xpadhT[((long)b * NPTS + n) * 16 + c] = f2bf(v);
}
// layer-1 weights: w1(64,6) -> padded, diff-transformed bf16 (64,32)
// y = wd.(x_nb - x_c) + wc.x_c  =  wd.x_nb + (wc - wd).x_c
__global__ void k_padw1(const float* __restrict__ w1, u16* __restrict__ wh) {
    int t = blockIdx.x * blockDim.x + threadIdx.x;
    if (t >= 64 * 32) return;
    int o = t / 32, c = t % 32;
    float v = 0.f;
    if (c < 3) v = w1[o * 6 + c];                                  // -> x_nb part
    else if (c >= 16 && c < 19)
        v = w1[o * 6 + 3 + (c - 16)] - w1[o * 6 + (c - 16)];       // -> x_ctr part
    wh[t] = f2bf(v);
}
// generic weight prep: fp32 (O,K) -> bf16; if C>0 apply diff transform on cols >= C
__global__ void k_prepw(const float* __restrict__ w, u16* __restrict__ wh,
                        int C, int K, int total) {
    int t = blockIdx.x * blockDim.x + threadIdx.x;
    if (t >= total) return;
    int o = t / K, c = t % K;
    float v = w[t];
    if (C > 0 && c >= C) v = v - w[o * K + c - C];
    wh[t] = f2bf(v);
}

// ---------------- squared norms: xx[b][n] = sum_c x^2 ----------------
__global__ void k_sqnorm(const float* __restrict__ in, long bstride, int C,
                         float* __restrict__ xx) {
    int t = blockIdx.x * blockDim.x + threadIdx.x;
    if (t >= BATCH * NPTS) return;
    int b = t / NPTS, n = t % NPTS;
    const float* p = in + (long)b * bstride + n;
    float s = 0.f;
    for (int c = 0; c < C; ++c) { float v = p[c * NPTS]; s += v * v; }
    xx[t] = s;
}

// ---------------- kNN top-20 via fp32 WMMA distance tiles ----------------
template <int C>
__global__ void __launch_bounds__(64) k_knn(const float* __restrict__ in,
                                            long bstride,
                                            const float* __restrict__ xx,
                                            int* __restrict__ idx) {
    static_assert(C % 4 == 0, "C must be multiple of 4");
    __shared__ float G[64][16];
    int b  = blockIdx.x / (NPTS / 64);
    int i0 = (blockIdx.x % (NPTS / 64)) * 64;
    int tid = threadIdx.x;
    int wave = tid >> 5, l = tid & 31, lm = l & 15;
    int khi = (l < 16) ? 0 : 2;                 // fp32 A/B K split across lane halves
    const float* xb = in + (long)b * bstride;

    float bestd[KNN]; int bestj[KNN];
#pragma unroll
    for (int t = 0; t < KNN; ++t) { bestd[t] = -3.4e38f; bestj[t] = 0; }
    float xxi = xx[b * NPTS + i0 + tid];

    for (int j0 = 0; j0 < NPTS; j0 += 16) {
        for (int st = wave * 2; st < wave * 2 + 2; ++st) {   // row sub-tiles
            int r0 = st * 16;
            v8f acc = {};
#pragma unroll
            for (int c0 = 0; c0 < C; c0 += 4) {
                v2f a, bb;
#pragma unroll
                for (int v = 0; v < 2; ++v) {
                    int c = c0 + khi + v;
                    a[v]  = xb[c * NPTS + i0 + r0 + lm];
                    bb[v] = xb[c * NPTS + j0 + lm];
                }
                acc = __builtin_amdgcn_wmma_f32_16x16x4_f32(
                        false, a, false, bb, (short)0, acc, false, false);
            }
#pragma unroll
            for (int v = 0; v < 8; ++v) {
                int m = v + ((l < 16) ? 0 : 8);
                G[r0 + m][lm] = acc[v];
            }
        }
        __syncthreads();
#pragma unroll 4
        for (int jj = 0; jj < 16; ++jj) {
            int j = j0 + jj;
            float d = 2.f * G[tid][jj] - xxi - xx[b * NPTS + j];
            if (d > bestd[KNN - 1]) {
                bestd[KNN - 1] = d; bestj[KNN - 1] = j;
#pragma unroll
                for (int t = KNN - 1; t > 0; --t) {
                    if (bestd[t] > bestd[t - 1]) {
                        float td = bestd[t]; bestd[t] = bestd[t - 1]; bestd[t - 1] = td;
                        int tj = bestj[t]; bestj[t] = bestj[t - 1]; bestj[t - 1] = tj;
                    }
                }
            }
        }
        __syncthreads();
    }
    int* op = idx + ((long)(b * NPTS + i0 + tid)) * KNN;
#pragma unroll
    for (int t = 0; t < KNN; ++t) op[t] = bestj[t];
}

__device__ __forceinline__ v8u load_afrag(const u16* wrow, int c0, int kbA) {
    uint4 a0 = *(const uint4*)&wrow[c0 + kbA];        // K kbA..kbA+7
    uint4 a1 = *(const uint4*)&wrow[c0 + kbA + 8];    // K kbA+8..kbA+15
    v8u av = { a0.x, a0.y, a0.z, a0.w, a1.x, a1.y, a1.z, a1.w };
    return av;
}

// ---------------- fused EdgeConv: bf16 WMMA + stats + max_k ---------------
// Y(O x N*20) = W'(O x 2C) * [x_nb ; x_ctr] (diff folded into W').
// Features stored point-major bf16 (B, N, rowlen): each lane's 16-element
// B run is 32 contiguous aligned bytes -> two b128 loads per K-step.
template <int C, int O>
__global__ void __launch_bounds__(128) k_edgeconv(
        const u16* __restrict__ xT, long rowlen, int cin0,
        const u16* __restrict__ wh, const int* __restrict__ idx,
        float* __restrict__ gsum, float* __restrict__ gssq,
        float* __restrict__ ymax) {
    constexpr int K2 = 2 * C;
    static_assert(K2 % 32 == 0, "2C must be multiple of 32");
    static_assert(C % 16 == 0, "C must be multiple of 16");
    constexpr int nOT = O / 16;
    __shared__ float s_sum[16], s_ssq[16];
    __shared__ unsigned s_max[64];                 // [o_local*4 + pt_local]
    int bid = blockIdx.x;
    int ptb = bid % (NPTS / 4);
    int ot  = (bid / (NPTS / 4)) % nOT;
    int b   = bid / ((NPTS / 4) * nOT);
    int tid = threadIdx.x;
    if (tid < 16) { s_sum[tid] = 0.f; s_ssq[tid] = 0.f; }
    if (tid < 64) s_max[tid] = 0u;
    __syncthreads();

    int wave = tid >> 5, l = tid & 31, lm = l & 15;
    const u16* xb = xT + (long)b * NPTS * rowlen;
    const u16* wrow = wh + (ot * 16 + lm) * K2;
    int p0 = ptb * 4;               // first point of this block
    int col0 = p0 * KNN;            // first edge-column
    int kbA = (l < 16) ? 0 : 16;    // 16-bit A layout K base
    int kbB = (l < 16) ? 0 : 16;    // 16-bit B layout K base

    for (int t = wave; t < 5; t += 4) {            // 5 col-tiles of 16 over 4 waves
        int j  = col0 + t * 16 + lm;               // edge column (pt, kk)
        int pt = j / KNN;
        int nb = idx[((long)b * NPTS + pt) * KNN + (j % KNN)];
        int ptl = pt - p0;
        const u16* rowNb = xb + (long)nb * rowlen + cin0;
        const u16* rowPt = xb + (long)pt * rowlen + cin0;
        v8f acc = {};
#pragma unroll
        for (int c0 = 0; c0 < K2; c0 += 32) {
            v8u av = load_afrag(wrow, c0, kbA);
            int cp0 = c0 + kbB;                    // run start; whole run same side
            const u16* src = (cp0 < C) ? (rowNb + cp0) : (rowPt + (cp0 - C));
            v16u bv = *(const v16u*)src;           // 32B contiguous -> 2x b128
            if (c0 + 32 < K2)
                __builtin_prefetch(src + 32, 0, 1);
            acc = __builtin_amdgcn_wmma_f32_16x16x32_bf16(
                    false, __builtin_bit_cast(v16bf, av),
                    false, __builtin_bit_cast(v16bf, bv),
                    (short)0, acc, false, false);
        }
#pragma unroll
        for (int v = 0; v < 8; ++v) {
            int ol = v + ((l < 16) ? 0 : 8);
            float y = acc[v];
            atomicAdd(&s_sum[ol], y);              // ds_add_f32
            atomicAdd(&s_ssq[ol], y * y);
            atomicMax(&s_max[ol * 4 + ptl], fmap(y));  // ds_max_u32
        }
    }
    __syncthreads();
    if (tid < 64) {
        int ol = tid >> 2, ptl = tid & 3;
        int og = ot * 16 + ol;
        ymax[((long)b * O + og) * NPTS + p0 + ptl] = funmap(s_max[tid]);
    }
    if (tid < 16) {
        int og = ot * 16 + tid;
        atomicAdd(&gsum[b * O + og], s_sum[tid]);
        atomicAdd(&gssq[b * O + og], s_ssq[tid]);
    }
}

// norm+leaky commute with max_k; dual-write fp32 (B,C,N) + bf16 (B,N,512)
__global__ void k_ec_finalize(const float* __restrict__ ymax,
                              const float* __restrict__ gsum,
                              const float* __restrict__ gssq,
                              int O, float* __restrict__ xcat,
                              u16* __restrict__ xcathT, int coff) {
    long t = (long)blockIdx.x * blockDim.x + threadIdx.x;
    long tot = (long)BATCH * O * NPTS;
    if (t >= tot) return;
    int n = (int)(t % NPTS); long r = t / NPTS;
    int o = (int)(r % O);    int b = (int)(r / O);
    float cnt  = (float)NPTS * (float)KNN;
    float mean = gsum[b * O + o] / cnt;
    float var  = gssq[b * O + o] / cnt - mean * mean;
    float z = (ymax[t] - mean) * rsqrtf(var + EPSV);
    z = (z >= 0.f) ? z : 0.2f * z;
    xcat[((long)b * 512 + coff + o) * NPTS + n] = z;
    xcathT[((long)b * NPTS + n) * 512 + coff + o] = f2bf(z);
}

// ---------------- classifier GEMM: y5 = w5(512x512) * xcat, bf16 WMMA ------
__global__ void __launch_bounds__(256) k_gemm5(
        const u16* __restrict__ xcathT, const u16* __restrict__ w5h,
        float* __restrict__ y5, float* __restrict__ g5sum,
        float* __restrict__ g5ssq) {
    __shared__ float s_sum[16], s_ssq[16];
    int bid = blockIdx.x;
    int nt = bid % (NPTS / 128);
    int ot = (bid / (NPTS / 128)) % 32;
    int b  = bid / ((NPTS / 128) * 32);
    int tid = threadIdx.x;
    if (tid < 16) { s_sum[tid] = 0.f; s_ssq[tid] = 0.f; }
    __syncthreads();
    int wave = tid >> 5, l = tid & 31, lm = l & 15;
    int n0 = nt * 128 + wave * 16;
    int o0 = ot * 16;
    const u16* xrow = xcathT + ((long)b * NPTS + n0 + lm) * 512;  // contiguous K
    const u16* wrow = w5h + (o0 + lm) * 512;
    int kbA = (l < 16) ? 0 : 16, kbB = (l < 16) ? 0 : 16;
    v8f acc = {};
#pragma unroll 4
    for (int c0 = 0; c0 < 512; c0 += 32) {
        v8u av = load_afrag(wrow, c0, kbA);
        v16u bv = *(const v16u*)&xrow[c0 + kbB];   // 32B contiguous -> 2x b128
        if (c0 + 32 < 512)
            __builtin_prefetch(&xrow[c0 + 32 + kbB], 0, 1);
        acc = __builtin_amdgcn_wmma_f32_16x16x32_bf16(
                false, __builtin_bit_cast(v16bf, av),
                false, __builtin_bit_cast(v16bf, bv),
                (short)0, acc, false, false);
    }
#pragma unroll
    for (int v = 0; v < 8; ++v) {
        int ol = v + ((l < 16) ? 0 : 8);
        float y = acc[v];
        y5[((long)b * 512 + o0 + ol) * NPTS + n0 + lm] = y;
        atomicAdd(&s_sum[ol], y);
        atomicAdd(&s_ssq[ol], y * y);
    }
    __syncthreads();
    if (tid < 16) {
        atomicAdd(&g5sum[o0 + tid], s_sum[tid]);
        atomicAdd(&g5ssq[o0 + tid], s_ssq[tid]);
    }
}

// ---------------- BN5 + leaky + max/mean heads ----------------
__global__ void k_bn5(const float* __restrict__ y5,
                      const float* __restrict__ g5sum,
                      const float* __restrict__ g5ssq,
                      const float* __restrict__ gamma,
                      const float* __restrict__ beta,
                      float* __restrict__ out) {
    int t = blockIdx.x * blockDim.x + threadIdx.x;
    if (t >= BATCH * 512) return;
    int o = t % 512, b = t / 512;
    float cnt  = (float)BATCH * (float)NPTS;
    float mean = g5sum[o] / cnt;
    float var  = g5ssq[o] / cnt - mean * mean;
    float rs   = rsqrtf(var + EPSV);
    float g = gamma[o], be = beta[o];
    const float* p = y5 + ((long)b * 512 + o) * NPTS;
    float mx = -3.4e38f, sm = 0.f;
    for (int n = 0; n < NPTS; ++n) {
        float z = (p[n] - mean) * rs * g + be;
        z = (z >= 0.f) ? z : 0.2f * z;
        mx = fmaxf(mx, z);
        sm += z;
    }
    out[b * 1024 + o]       = mx;
    out[b * 1024 + 512 + o] = sm / (float)NPTS;
}

// ---------------- per-layer driver (template dispatch) ----------------
template <int C, int O>
static void run_layer(const float* in, long bstride,
                      const u16* xT, long rowlen, int cin0,
                      const u16* wh, int coff,
                      float* xx, int* idx, float* gsum, float* gssq,
                      float* ymax, float* xcat, u16* xcathT, hipStream_t stream) {
    hipMemsetAsync(gsum, 0, 2 * BATCH * 512 * sizeof(float), stream);
    k_sqnorm<<<(BATCH * NPTS + 255) / 256, 256, 0, stream>>>(in, bstride, C, xx);
    k_knn<C><<<BATCH * (NPTS / 64), 64, 0, stream>>>(in, bstride, xx, idx);
    k_edgeconv<C, O><<<BATCH * (O / 16) * (NPTS / 4), 128, 0, stream>>>(
        xT, rowlen, cin0, wh, idx, gsum, gssq, ymax);
    long tot = (long)BATCH * O * NPTS;
    k_ec_finalize<<<(unsigned)((tot + 255) / 256), 256, 0, stream>>>(
        ymax, gsum, gssq, O, xcat, xcathT, coff);
}

extern "C" void kernel_launch(void* const* d_in, const int* in_sizes, int n_in,
                              void* d_out, int out_size, void* d_ws, size_t ws_size,
                              hipStream_t stream) {
    (void)in_sizes; (void)n_in; (void)out_size; (void)ws_size;
    const float* x   = (const float*)d_in[0];
    const float* w1  = (const float*)d_in[1];
    const float* w2  = (const float*)d_in[2];
    const float* w3  = (const float*)d_in[3];
    const float* w4  = (const float*)d_in[4];
    const float* w5  = (const float*)d_in[5];
    const float* gam = (const float*)d_in[6];
    const float* bet = (const float*)d_in[7];

    // workspace layout:
    float* xcat   = (float*)d_ws;                             // 16*512*2048 f32 (B,C,N)
    float* y5     = xcat + (long)BATCH * 512 * NPTS;          // 16*512*2048 f32 (also ymax)
    float* xx     = y5 + (long)BATCH * 512 * NPTS;            // 16*2048
    int*   idx    = (int*)(xx + BATCH * NPTS);                // 16*2048*20
    float* gsum   = (float*)(idx + (long)BATCH * NPTS * KNN); // 16*512
    float* gssq   = gsum + BATCH * 512;                       // 16*512
    float* xpad   = gssq + BATCH * 512;                       // 16*16*2048 f32 (B,C,N)
    u16*   xcathT = (u16*)(xpad + (long)BATCH * 16 * NPTS);   // 16*2048*512 bf16 (B,N,C)
    u16*   xpadhT = xcathT + (long)BATCH * NPTS * 512;        // 16*2048*16 bf16 (B,N,C)
    u16*   wh1    = xpadhT + (long)BATCH * NPTS * 16;         // 64*32
    u16*   wh2    = wh1 + 64 * 32;                            // 64*128
    u16*   wh3    = wh2 + 64 * 128;                           // 128*128
    u16*   wh4    = wh3 + 128 * 128;                          // 256*256
    u16*   w5h    = wh4 + 256 * 256;                          // 512*512

    // prep: zero-pad layer 1, diff-transform + bf16-convert all weights
    k_padx<<<(unsigned)(((long)BATCH * 16 * NPTS + 255) / 256), 256, 0, stream>>>(
        x, xpad, xpadhT);
    k_padw1<<<(64 * 32 + 255) / 256, 256, 0, stream>>>(w1, wh1);
    k_prepw<<<(64 * 128 + 255) / 256, 256, 0, stream>>>(w2, wh2, 64, 128, 64 * 128);
    k_prepw<<<(128 * 128 + 255) / 256, 256, 0, stream>>>(w3, wh3, 64, 128, 128 * 128);
    k_prepw<<<(256 * 256 + 255) / 256, 256, 0, stream>>>(w4, wh4, 128, 256, 256 * 256);
    k_prepw<<<(512 * 512 + 255) / 256, 256, 0, stream>>>(w5, w5h, 0, 512, 512 * 512);

    run_layer<16,  64>(xpad, 16L * NPTS, xpadhT, 16, 0, wh1, 0,
                       xx, idx, gsum, gssq, y5, xcat, xcathT, stream);
    run_layer<64,  64>(xcat + 0L * NPTS, 512L * NPTS, xcathT, 512, 0, wh2, 64,
                       xx, idx, gsum, gssq, y5, xcat, xcathT, stream);
    run_layer<64, 128>(xcat + 64L * NPTS, 512L * NPTS, xcathT, 512, 64, wh3, 128,
                       xx, idx, gsum, gssq, y5, xcat, xcathT, stream);
    run_layer<128,256>(xcat + 128L * NPTS, 512L * NPTS, xcathT, 512, 128, wh4, 256,
                       xx, idx, gsum, gssq, y5, xcat, xcathT, stream);

    hipMemsetAsync(gsum, 0, 2 * BATCH * 512 * sizeof(float), stream);
    k_gemm5<<<BATCH * 32 * (NPTS / 128), 256, 0, stream>>>(xcathT, w5h, y5, gsum, gssq);
    k_bn5<<<(BATCH * 512 + 255) / 256, 256, 0, stream>>>(
        y5, gsum, gssq, gam, bet, (float*)d_out);
}
